// S4Deter_18373870092480
// MI455X (gfx1250) — matrix operand order repaired and checked by reading, not compiled, not run on hardware
//
#include <hip/hip_runtime.h>

// ---------------------------------------------------------------------------
// MI455X (gfx1250) fused S4 world-model step.
// bf16 WMMA (v_wmma_f32_16x16x32_bf16) for all GEMMs, f32 residual stream.
// One fused kernel per S4 layer: rmsnorm -> GEMM(D->S) -> in-register decay
// scan over the 16 tokens -> GEMM(S->D) + skip GEMM(D->D) -> residual add.
// Register-pressure shaped so every scheduling region stays inside the
// direct 256-VGPR window (no s_set_vgpr_msb, no acc v_mov, no hazard NOPs):
//   - k_layer: two M-tiles share B frags, but N-tiles processed in 2 groups
//     of 4 (64 live accumulators per region, #pragma unroll 1 between).
//   - k_input_proj: single M-tile per wave (64 accumulators).
// ---------------------------------------------------------------------------

typedef __attribute__((ext_vector_type(16))) __bf16   v16bf;
typedef __attribute__((ext_vector_type(8)))  float    v8f;
typedef __attribute__((ext_vector_type(4)))  float    v4f;
typedef __attribute__((ext_vector_type(4)))  unsigned v4u;

union FragAB { v4u q[2]; v16bf v; };          // 32 bytes: one WMMA A or B frag
union Pack8  { __bf16 h[8]; v4u q; };         // 8 bf16 packed for a b128 store

__device__ __attribute__((always_inline)) inline v8f v8f_zero() {
  v8f v;
#pragma unroll
  for (int i = 0; i < 8; ++i) v[i] = 0.f;
  return v;
}

// One wave: two 16-row M-tiles (A rows xrow0/xrow1 in LDS) x FOUR N-tiles
// starting at nbase. B rows are bf16 weight rows [N][K=256], contiguous in K
// (L2 resident), shared by both M-tiles.
__device__ __attribute__((always_inline)) inline void gemm_tiles2g(
    const __bf16* __restrict__ xrow0, const __bf16* __restrict__ xrow1,
    const __bf16* __restrict__ W, int nbase, int n16, int hlf,
    v8f acc0[4], v8f acc1[4])
{
  for (int kc = 0; kc < 256; kc += 32) {
    FragAB a0, a1;
    const __bf16* pa0 = xrow0 + kc + hlf * 8;   // A: K = 16*(v/4)+8*half+2*(v%4)
    const __bf16* pa1 = xrow1 + kc + hlf * 8;
    a0.q[0] = *(const v4u*)pa0;  a0.q[1] = *(const v4u*)(pa0 + 16);
    a1.q[0] = *(const v4u*)pa1;  a1.q[1] = *(const v4u*)(pa1 + 16);
#pragma unroll
    for (int i = 0; i < 4; ++i) {
      const __bf16* pb = W + (size_t)((nbase + i) * 16 + n16) * 256 + kc + hlf * 8;
      FragAB bf_;
      bf_.q[0] = *(const v4u*)pb;               // B: K = 8*half + 0..7, 16 + 8*half + 0..7
      bf_.q[1] = *(const v4u*)(pb + 16);
      acc0[i] = __builtin_amdgcn_wmma_f32_16x16x32_bf16(
          false, a0.v, false, bf_.v, (short)0, acc0[i], false, false);
      acc1[i] = __builtin_amdgcn_wmma_f32_16x16x32_bf16(
          false, a1.v, false, bf_.v, (short)0, acc1[i], false, false);
    }
  }
}

// ------------------------------- prep kernels ------------------------------

__global__ void k_cvt_bf16(const float* __restrict__ s, __bf16* __restrict__ d, int n) {
  int i = blockIdx.x * blockDim.x + threadIdx.x;
  if (i < n) d[i] = (__bf16)s[i];
}

__global__ void k_decay(const float* __restrict__ ld, float* __restrict__ d, int n) {
  int i = blockIdx.x * blockDim.x + threadIdx.x;
  if (i < n) d[i] = 1.f / (1.f + __expf(-ld[i]));
}

// hcat[b][0:1024] = stoch, hcat[b][1024:1056] = action / max(|action|,1), bf16
__global__ void k_hcat(const float* __restrict__ st, const float* __restrict__ act,
                       __bf16* __restrict__ d, int n) {
  int i = blockIdx.x * blockDim.x + threadIdx.x;
  if (i >= n) return;
  int b = i / 1056, j = i - b * 1056;
  float v;
  if (j < 1024) v = st[(size_t)b * 1024 + j];
  else { float a = act[(size_t)b * 32 + (j - 1024)]; v = a / fmaxf(fabsf(a), 1.f); }
  d[i] = (__bf16)v;
}

// ------------------------------ input projection ---------------------------
// h = hcat @ W0^T + b0 ; new_tok = silu(rmsnorm(h, g0)).  64 rows per block,
// one 16-row M-tile per wave (low register pressure -> clean pipelined loop).
__global__ __launch_bounds__(256) void k_input_proj(
    const __bf16* __restrict__ hcat, const __bf16* __restrict__ w0b,
    const float* __restrict__ b0, const float* __restrict__ g0,
    float* __restrict__ newtok)
{
  __shared__ float wsum[8][16];
  const int tid = threadIdx.x, w = tid >> 5, lane = tid & 31;
  const int hlf = lane >> 4, n16 = lane & 15;
  const int mb = w >> 1, nsel = w & 1;
  const int m0 = blockIdx.x * 64 + mb * 16;

  v8f acc[8];
#pragma unroll
  for (int i = 0; i < 8; ++i) acc[i] = v8f_zero();

  const __bf16* arow = hcat + (size_t)(m0 + n16) * 1056;
  for (int kc = 0; kc < 1056; kc += 32) {
    FragAB a;
    const __bf16* pa = arow + kc + hlf * 8;
    a.q[0] = *(const v4u*)pa;
    a.q[1] = *(const v4u*)(pa + 16);
#pragma unroll
    for (int i = 0; i < 8; ++i) {
      const __bf16* pb = w0b + (size_t)((nsel * 8 + i) * 16 + n16) * 1056 + kc + hlf * 8;
      FragAB bf_;
      bf_.q[0] = *(const v4u*)pb;
      bf_.q[1] = *(const v4u*)(pb + 16);
      acc[i] = __builtin_amdgcn_wmma_f32_16x16x32_bf16(
          false, a.v, false, bf_.v, (short)0, acc[i], false, false);
    }
  }

  // + b0, then per-row sum of squares (this wave owns 128 of the 256 cols)
  float ss[8];
#pragma unroll
  for (int r = 0; r < 8; ++r) ss[r] = 0.f;
#pragma unroll
  for (int i = 0; i < 8; ++i) {
    int d = (nsel * 8 + i) * 16 + n16;
    float bv = b0[d];
#pragma unroll
    for (int r = 0; r < 8; ++r) { acc[i][r] += bv; ss[r] += acc[i][r] * acc[i][r]; }
  }
#pragma unroll
  for (int m = 1; m < 16; m <<= 1) {
#pragma unroll
    for (int r = 0; r < 8; ++r) ss[r] += __shfl_xor(ss[r], m, 32);
  }
  if (n16 == 0) {
#pragma unroll
    for (int r = 0; r < 8; ++r) wsum[w][hlf * 8 + r] = ss[r];
  }
  __syncthreads();
  float sc[8];
#pragma unroll
  for (int r = 0; r < 8; ++r) {
    float tot = ss[r] + wsum[w ^ 1][hlf * 8 + r];   // partner: other 128 cols
    sc[r] = rsqrtf(tot * (1.f / 256.f) + 1e-4f);
  }
#pragma unroll
  for (int i = 0; i < 8; ++i) {
    int d = (nsel * 8 + i) * 16 + n16;
    float g = g0[d];
#pragma unroll
    for (int r = 0; r < 8; ++r) {
      float x = acc[i][r] * sc[r] * g;
      float y = x / (1.f + __expf(-x));             // silu
      newtok[(size_t)(m0 + r + 8 * hlf) * 256 + d] = y;
    }
  }
}

// ------------------------------- fused S4 layer ----------------------------
// Block = 8 batch rows x 16 tokens (128 M-rows) x full D=256.
__global__ __launch_bounds__(256) void k_layer(
    const float* __restrict__ src, const float* __restrict__ deter,
    const float* __restrict__ newtok, int first,
    const __bf16* __restrict__ Wi, const __bf16* __restrict__ Wo,
    const __bf16* __restrict__ Wsk,
    const float* __restrict__ bi, const float* __restrict__ bo,
    const float* __restrict__ bsk,
    const float* __restrict__ nw, const float* __restrict__ decay,
    float* __restrict__ dst)
{
  __shared__ __bf16 xs[128][264];                 // 67.6 KB normalized tokens
  __shared__ __bf16 st[128][264];                 // 67.6 KB scan states
  const int tid = threadIdx.x, w = tid >> 5, lane = tid & 31;
  const int hlf = lane >> 4, n16 = lane & 15;
  const int b0i = blockIdx.x * 8;

  // warm weight rows into cache (global_prefetch_b8)
  if (tid < 128) {
    __builtin_prefetch(Wi  + (size_t)tid * 512, 0, 1);
    __builtin_prefetch(Wo  + (size_t)tid * 512, 0, 1);
    __builtin_prefetch(Wsk + (size_t)tid * 512, 0, 1);
  }

  // Phase 1: rmsnorm(tokens) -> xs (bf16). Each wave: one batch row's 16 tokens.
  {
    int b = b0i + w;
    for (int t = 0; t < 16; ++t) {
      const float* p;
      if (first) p = (t < 15) ? (deter + (size_t)b * 4096 + (t + 1) * 256)
                              : (newtok + (size_t)b * 256);
      else       p = src + (size_t)b * 4096 + t * 256;
      v4f v0 = ((const v4f*)p)[lane * 2];
      v4f v1 = ((const v4f*)p)[lane * 2 + 1];
      float ssq = v0.x*v0.x + v0.y*v0.y + v0.z*v0.z + v0.w*v0.w
                + v1.x*v1.x + v1.y*v1.y + v1.z*v1.z + v1.w*v1.w;
#pragma unroll
      for (int m = 1; m < 32; m <<= 1) ssq += __shfl_xor(ssq, m, 32);
      float scl = rsqrtf(ssq * (1.f / 256.f) + 1e-4f);
      int c = lane * 8;
      Pack8 pk;
      pk.h[0]=(__bf16)(v0.x*scl*nw[c+0]); pk.h[1]=(__bf16)(v0.y*scl*nw[c+1]);
      pk.h[2]=(__bf16)(v0.z*scl*nw[c+2]); pk.h[3]=(__bf16)(v0.w*scl*nw[c+3]);
      pk.h[4]=(__bf16)(v1.x*scl*nw[c+4]); pk.h[5]=(__bf16)(v1.y*scl*nw[c+5]);
      pk.h[6]=(__bf16)(v1.z*scl*nw[c+6]); pk.h[7]=(__bf16)(v1.w*scl*nw[c+7]);
      *(v4u*)&xs[w * 16 + t][c] = pk.q;
    }
  }
  __syncthreads();

  const int nsel = w & 1;
  const int mb0 = w >> 1, mb1 = mb0 + 4;          // two M-tiles per wave
  const __bf16* xrow0 = &xs[mb0 * 16 + n16][0];
  const __bf16* xrow1 = &xs[mb1 * 16 + n16][0];

  // Phase 2: u = x @ Wi^T + bi, decay scan over t in the C-tile regs, states
  // (bf16) -> st buffer. Two groups of 4 N-tiles keep live accs at 64.
#pragma unroll 1
  for (int g = 0; g < 2; ++g) {
    const int nbase = nsel * 8 + g * 4;
    v8f u0[4], u1[4];
#pragma unroll
    for (int i = 0; i < 4; ++i) { u0[i] = v8f_zero(); u1[i] = v8f_zero(); }
    gemm_tiles2g(xrow0, xrow1, Wi, nbase, n16, hlf, u0, u1);
#pragma unroll
    for (int i = 0; i < 4; ++i) {
      int s = (nbase + i) * 16 + n16;
      float bv = bi[s], dv = decay[s];
#pragma unroll
      for (int r = 0; r < 8; ++r) { u0[i][r] += bv; u1[i][r] += bv; }
      // local prefix over this half's 8 tokens (C layout: VGPR r <-> t=r+8*half)
#pragma unroll
      for (int r = 1; r < 8; ++r) {
        u0[i][r] += dv * u0[i][r - 1];
        u1[i][r] += dv * u1[i][r - 1];
      }
      // carry state(t=7) from lanes 0-15 into lanes 16-31, scaled by decay^(r+1)
      float c0 = __shfl(u0[i][7], n16, 32);
      float c1 = __shfl(u1[i][7], n16, 32);
      if (hlf) {
        float f = dv;
#pragma unroll
        for (int r = 0; r < 8; ++r) { u0[i][r] += f * c0; u1[i][r] += f * c1; f *= dv; }
      }
      // states -> st (xs stays intact for the skip GEMM)
#pragma unroll
      for (int r = 0; r < 8; ++r) {
        st[mb0 * 16 + r + 8 * hlf][s] = (__bf16)u0[i][r];
        st[mb1 * 16 + r + 8 * hlf][s] = (__bf16)u1[i][r];
      }
    }
  }
  __syncthreads();                                // states visible to all waves

  // Phase 3: out = x @ Ws^T + states @ Wo^T, residual epilogue.
  // Same 2-group structure: one 64-reg accumulator set live at a time.
  const __bf16* srow0 = &st[mb0 * 16 + n16][0];
  const __bf16* srow1 = &st[mb1 * 16 + n16][0];
#pragma unroll 1
  for (int g = 0; g < 2; ++g) {
    const int nbase = nsel * 8 + g * 4;
    v8f o0[4], o1[4];
#pragma unroll
    for (int i = 0; i < 4; ++i) { o0[i] = v8f_zero(); o1[i] = v8f_zero(); }
    gemm_tiles2g(xrow0, xrow1, Wsk, nbase, n16, hlf, o0, o1);
    gemm_tiles2g(srow0, srow1, Wo,  nbase, n16, hlf, o0, o1);
#pragma unroll
    for (int i = 0; i < 4; ++i) {
      int d = (nbase + i) * 16 + n16;
      float bb = bo[d] + bsk[d];
#pragma unroll
      for (int r = 0; r < 8; ++r) {
        int t = r + 8 * hlf;
        int ba = b0i + mb0, bb2 = b0i + mb1;
        float res0, res1;
        if (first) {
          res0 = (t < 15) ? deter[(size_t)ba * 4096 + (t + 1) * 256 + d]
                          : newtok[(size_t)ba * 256 + d];
          res1 = (t < 15) ? deter[(size_t)bb2 * 4096 + (t + 1) * 256 + d]
                          : newtok[(size_t)bb2 * 256 + d];
        } else {
          res0 = src[(size_t)ba * 4096 + t * 256 + d];
          res1 = src[(size_t)bb2 * 4096 + t * 256 + d];
        }
        dst[(size_t)ba * 4096 + t * 256 + d]  = res0 + o0[i][r] + bb;
        dst[(size_t)bb2 * 4096 + t * 256 + d] = res1 + o1[i][r] + bb;
      }
    }
  }
}

// ------------------------------- final rmsnorm -----------------------------
__global__ __launch_bounds__(256) void k_outnorm(
    const float* __restrict__ src, const float* __restrict__ onw,
    float* __restrict__ out)
{
  int row = blockIdx.x * 8 + (threadIdx.x >> 5);
  int lane = threadIdx.x & 31;
  const float* p = src + (size_t)row * 256 + lane * 8;
  v4f v0 = *(const v4f*)p, v1 = *(const v4f*)(p + 4);
  float ssq = v0.x*v0.x + v0.y*v0.y + v0.z*v0.z + v0.w*v0.w
            + v1.x*v1.x + v1.y*v1.y + v1.z*v1.z + v1.w*v1.w;
#pragma unroll
  for (int m = 1; m < 32; m <<= 1) ssq += __shfl_xor(ssq, m, 32);
  float sc = rsqrtf(ssq * (1.f / 256.f) + 1e-4f);
  float* q = out + (size_t)row * 256 + lane * 8;
  const float* g = onw + lane * 8;
  q[0]=v0.x*sc*g[0]; q[1]=v0.y*sc*g[1]; q[2]=v0.z*sc*g[2]; q[3]=v0.w*sc*g[3];
  q[4]=v1.x*sc*g[4]; q[5]=v1.y*sc*g[5]; q[6]=v1.z*sc*g[6]; q[7]=v1.w*sc*g[7];
}

// ---------------------------------------------------------------------------

extern "C" void kernel_launch(void* const* d_in, const int* in_sizes, int n_in,
                              void* d_out, int out_size, void* d_ws, size_t ws_size,
                              hipStream_t stream) {
  (void)in_sizes; (void)n_in; (void)out_size; (void)ws_size;
  const float* stoch     = (const float*)d_in[0];
  const float* deter     = (const float*)d_in[1];
  const float* action    = (const float*)d_in[2];
  const float* W0        = (const float*)d_in[3];
  const float* b0        = (const float*)d_in[4];
  const float* g0        = (const float*)d_in[5];
  const float* norm_w    = (const float*)d_in[6];
  const float* W_in      = (const float*)d_in[7];
  const float* b_in      = (const float*)d_in[8];
  const float* W_out     = (const float*)d_in[9];
  const float* b_out     = (const float*)d_in[10];
  const float* W_skip    = (const float*)d_in[11];
  const float* b_skip    = (const float*)d_in[12];
  const float* log_decay = (const float*)d_in[13];
  const float* out_nw    = (const float*)d_in[14];

  char* base = (char*)d_ws; size_t off = 0;
  auto alloc = [&](size_t bytes) -> void* {
    void* p = base + off; off = (off + bytes + 255) & ~(size_t)255; return p;
  };
  const int B = 4096, K0 = 1056;
  __bf16* hcat  = (__bf16*)alloc((size_t)B * K0 * 2);
  __bf16* w0b   = (__bf16*)alloc((size_t)256 * K0 * 2);
  __bf16* wib   = (__bf16*)alloc((size_t)4 * 256 * 256 * 2);
  __bf16* wob   = (__bf16*)alloc((size_t)4 * 256 * 256 * 2);
  __bf16* wsb   = (__bf16*)alloc((size_t)4 * 256 * 256 * 2);
  float*  dcy   = (float*)alloc((size_t)4 * 256 * 4);
  float*  ntok  = (float*)alloc((size_t)B * 256 * 4);
  float*  tokA  = (float*)alloc((size_t)B * 4096 * 4);
  float*  tokB  = (float*)alloc((size_t)B * 4096 * 4);

  // prep: bf16 weights, decay, concatenated bf16 input
  k_cvt_bf16<<<(256 * K0 + 255) / 256, 256, 0, stream>>>(W0, w0b, 256 * K0);
  k_cvt_bf16<<<(4 * 65536 + 255) / 256, 256, 0, stream>>>(W_in, wib, 4 * 65536);
  k_cvt_bf16<<<(4 * 65536 + 255) / 256, 256, 0, stream>>>(W_out, wob, 4 * 65536);
  k_cvt_bf16<<<(4 * 65536 + 255) / 256, 256, 0, stream>>>(W_skip, wsb, 4 * 65536);
  k_decay<<<(1024 + 255) / 256, 256, 0, stream>>>(log_decay, dcy, 1024);
  k_hcat<<<((B * K0) + 255) / 256, 256, 0, stream>>>(stoch, action, hcat, B * K0);

  // input projection -> new token
  k_input_proj<<<B / 64, 256, 0, stream>>>(hcat, w0b, b0, g0, ntok);

  // 4 fused S4 layers (ping-pong f32 token stream, L2 resident)
  float* bufs[2] = { tokA, tokB };
  const float* cur = nullptr;
  for (int l = 0; l < 4; ++l) {
    float* dst = bufs[l & 1];
    k_layer<<<B / 8, 256, 0, stream>>>(
        cur, deter, ntok, (l == 0) ? 1 : 0,
        wib + (size_t)l * 65536, wob + (size_t)l * 65536, wsb + (size_t)l * 65536,
        b_in + l * 256, b_out + l * 256, b_skip + l * 256,
        norm_w + l * 256, dcy + l * 256, dst);
    cur = dst;
  }

  // final rmsnorm -> d_out
  k_outnorm<<<(B * 16) / 8, 256, 0, stream>>>(cur, out_nw, (float*)d_out);
}
